// PointConvBNPositionalEncoding_14070312862123
// MI455X (gfx1250) — matrix operand order repaired and checked by reading, not compiled, not run on hardware
//
#include <hip/hip_runtime.h>
#include <hip/hip_bf16.h>

typedef __attribute__((ext_vector_type(16))) _Float16 v16h;
typedef __attribute__((ext_vector_type(8)))  _Float16 v8h;
typedef __attribute__((ext_vector_type(8)))  float    v8f;

#define BN_EPS 1e-5f
#define COUT   288
#define ROWS   64       // rows per block in the fused kernel
#define LDA    296      // LDS row stride in halves (288 + 8 pad), 16B aligned

// ---------------------------------------------------------------------------
// Kernel 1: 14-float reduction over xyz: sum(x_k) [4] and sum(x_j*x_k) [10]
// ---------------------------------------------------------------------------
__global__ __launch_bounds__(256) void stats_kernel(const float4* __restrict__ xyz,
                                                    float* __restrict__ stats, int N) {
    float s[14];
#pragma unroll
    for (int j = 0; j < 14; ++j) s[j] = 0.f;

    int stride = gridDim.x * blockDim.x;
    for (int i = blockIdx.x * blockDim.x + threadIdx.x; i < N; i += stride) {
        float4 p = xyz[i];
        s[0] += p.x; s[1] += p.y; s[2] += p.z; s[3] += p.w;
        s[4]  += p.x * p.x; s[5]  += p.x * p.y; s[6]  += p.x * p.z; s[7] += p.x * p.w;
        s[8]  += p.y * p.y; s[9]  += p.y * p.z; s[10] += p.y * p.w;
        s[11] += p.z * p.z; s[12] += p.z * p.w; s[13] += p.w * p.w;
    }

    // wave32 reduction
#pragma unroll
    for (int j = 0; j < 14; ++j) {
        for (int off = 16; off > 0; off >>= 1)
            s[j] += __shfl_down(s[j], off, 32);
    }

    __shared__ float red[8][14];
    int lane = threadIdx.x & 31, wave = threadIdx.x >> 5;
    if (lane == 0) {
#pragma unroll
        for (int j = 0; j < 14; ++j) red[wave][j] = s[j];
    }
    __syncthreads();
    if (threadIdx.x == 0) {
#pragma unroll
        for (int j = 0; j < 14; ++j) {
            float t = red[0][j];
            for (int w = 1; w < 8; ++w) t += red[w][j];
            atomicAdd(&stats[j], t);
        }
    }
}

// ---------------------------------------------------------------------------
// Kernel 2: fold BN into W1/b1 (analytic mean/var from 4x4 moments) and
//           convert W2 to f16 row-major for WMMA B-fragments.
// ---------------------------------------------------------------------------
__global__ __launch_bounds__(COUT) void prep_kernel(const float* __restrict__ W1,
                                                    const float* __restrict__ b1,
                                                    const float* __restrict__ gamma,
                                                    const float* __restrict__ beta,
                                                    const float* __restrict__ W2,
                                                    const float* __restrict__ stats, int N,
                                                    float* __restrict__ W1f,
                                                    float* __restrict__ b1f,
                                                    _Float16* __restrict__ W2h) {
    int c = threadIdx.x;
    float invN = 1.0f / (float)N;
    if (c < COUT) {
        float mu0 = stats[0] * invN, mu1 = stats[1] * invN;
        float mu2 = stats[2] * invN, mu3 = stats[3] * invN;
        float S00 = stats[4]  * invN, S01 = stats[5]  * invN, S02 = stats[6]  * invN, S03 = stats[7] * invN;
        float S11 = stats[8]  * invN, S12 = stats[9]  * invN, S13 = stats[10] * invN;
        float S22 = stats[11] * invN, S23 = stats[12] * invN, S33 = stats[13] * invN;

        float w0 = W1[c * 4 + 0], w1 = W1[c * 4 + 1];
        float w2 = W1[c * 4 + 2], w3 = W1[c * 4 + 3];
        float bc = b1[c];

        float wm = w0 * mu0 + w1 * mu1 + w2 * mu2 + w3 * mu3;
        float m  = wm + bc;
        float q  = w0 * w0 * S00 + w1 * w1 * S11 + w2 * w2 * S22 + w3 * w3 * S33
                 + 2.f * (w0 * w1 * S01 + w0 * w2 * S02 + w0 * w3 * S03 +
                          w1 * w2 * S12 + w1 * w3 * S13 + w2 * w3 * S23);
        float E2  = q + 2.f * bc * wm + bc * bc;
        float var = E2 - m * m;
        float a   = gamma[c] * rsqrtf(var + BN_EPS);

        W1f[c * 4 + 0] = a * w0;
        W1f[c * 4 + 1] = a * w1;
        W1f[c * 4 + 2] = a * w2;
        W1f[c * 4 + 3] = a * w3;
        b1f[c] = a * (bc - m) + beta[c];
    }
    for (int i = threadIdx.x; i < COUT * COUT; i += blockDim.x)
        W2h[i] = (_Float16)W2[i];
}

// ---------------------------------------------------------------------------
// Kernel 3: fused  A = f16(relu(xyz @ W1f^T + b1f))  ->  out = A @ W2h^T + b2
// block = 288 threads (9 waves), 64 rows/block.
// Per wave: ct2 outer (unroll 1, preload 9 B-fragments = 72 VGPRs, reused 4x),
// rt inner (unroll 1), kt chain of 9 v_wmma_f32_16x16x32_f16 unrolled.
// b2 folded into accumulator init; stores use 32-bit indexing with a single
// block-level bounds branch (only the last block is partial).
// ---------------------------------------------------------------------------
__global__ __launch_bounds__(COUT) void fused_kernel(const float* __restrict__ xyz,
                                                     const float* __restrict__ W1f,
                                                     const float* __restrict__ b1f,
                                                     const _Float16* __restrict__ W2h,
                                                     const float* __restrict__ b2,
                                                     float* __restrict__ out, int N) {
    __shared__ float xs[ROWS * 4];
    __shared__ __align__(16) _Float16 A[ROWS * LDA];

    int tid  = threadIdx.x;
    int base = blockIdx.x * ROWS;               // N <= 2^31/288, fits int
    bool fullblk = (base + ROWS) <= N;

    // stage xyz tile (64 rows x 4) into LDS; OOB rows -> 0
    if (tid < ROWS * 4) {
        int r = tid >> 2, comp = tid & 3;
        int g = base + r;
        xs[tid] = (g < N) ? xyz[g * 4 + comp] : 0.f;
    }
    __syncthreads();

    // fold + relu + f16 -> A tile; thread t owns channel c = t for all 64 rows
    {
        int c = tid;
        float w0 = W1f[c * 4 + 0], w1 = W1f[c * 4 + 1];
        float w2 = W1f[c * 4 + 2], w3 = W1f[c * 4 + 3];
        float bb = b1f[c];
#pragma unroll 4
        for (int r = 0; r < ROWS; ++r) {
            float h = fmaf(xs[r * 4 + 0], w0,
                      fmaf(xs[r * 4 + 1], w1,
                      fmaf(xs[r * 4 + 2], w2,
                      fmaf(xs[r * 4 + 3], w3, bb))));
            h = fmaxf(h, 0.f);
            A[r * LDA + c] = (_Float16)h;
        }
    }
    __syncthreads();

    int lane = tid & 31;
    int wave = tid >> 5;        // 0..8
    int half = lane >> 4;       // 0 for lanes 0-15, 1 for lanes 16-31
    int l16  = lane & 15;

#pragma unroll 1
    for (int ct2 = 0; ct2 < 2; ++ct2) {
        int ct  = wave * 2 + ct2;       // col tile 0..17
        int col = ct * 16 + l16;
        float bias = b2[col];

        // Preload all 9 B fragments for this column tile (reused by 4 row tiles).
        // 16-bit B 32x16 layout: halves 0-7 = K k0+8*half+i,
        // halves 8-15 = K k0+16+8*half+i, for column n = l16.
        const _Float16* Brow = W2h + col * COUT;
        v16h bv[9];
#pragma unroll
        for (int kt = 0; kt < 9; ++kt) {
            int k0 = kt * 32;
            v8h blo = *(const v8h*)(Brow + k0 + half * 8);
            v8h bhi = *(const v8h*)(Brow + k0 + 16 + half * 8);
#pragma unroll
            for (int i = 0; i < 8; ++i) { bv[kt][i] = blo[i]; bv[kt][8 + i] = bhi[i]; }
        }

#pragma unroll 1
        for (int rt = 0; rt < 4; ++rt) {
            v8f acc;
#pragma unroll
            for (int i = 0; i < 8; ++i) acc[i] = bias;   // fold b2 into C

            const _Float16* Arow = &A[(rt * 16 + l16) * LDA];
#pragma unroll
            for (int kt = 0; kt < 9; ++kt) {
                int k0 = kt * 32;
                // 16-bit A 16x32 layout: halves 0-7 = K k0+8*half+i,
                // halves 8-15 = K k0+16+8*half+i
                v8h alo = *(const v8h*)(Arow + k0 + half * 8);
                v8h ahi = *(const v8h*)(Arow + k0 + 16 + half * 8);
                v16h av;
#pragma unroll
                for (int i = 0; i < 8; ++i) { av[i] = alo[i]; av[8 + i] = ahi[i]; }
                acc = __builtin_amdgcn_wmma_f32_16x16x32_f16(
                    /*neg_a=*/false, av, /*neg_b=*/false, bv[kt],
                    /*c_mod=*/(short)0, acc, /*reuse_a=*/false, /*reuse_b=*/false);
            }

            // C/D layout: VGPR v -> M = v + 8*half, N = l16
            int row0 = base + rt * 16 + half * 8;
            float* p = out + row0 * COUT + col;          // 32-bit index math
            if (fullblk) {
#pragma unroll
                for (int v = 0; v < 8; ++v) p[v * COUT] = acc[v];
            } else {
#pragma unroll
                for (int v = 0; v < 8; ++v)
                    if (row0 + v < N) p[v * COUT] = acc[v];
            }
        }
    }
}

// ---------------------------------------------------------------------------
extern "C" void kernel_launch(void* const* d_in, const int* in_sizes, int n_in,
                              void* d_out, int out_size, void* d_ws, size_t ws_size,
                              hipStream_t stream) {
    const float* xyz   = (const float*)d_in[0];
    const float* W1    = (const float*)d_in[1];
    const float* b1    = (const float*)d_in[2];
    const float* gamma = (const float*)d_in[3];
    const float* beta  = (const float*)d_in[4];
    const float* W2    = (const float*)d_in[5];
    const float* b2    = (const float*)d_in[6];
    int N = in_sizes[0] / 4;

    char* ws = (char*)d_ws;
    float*    stats = (float*)(ws + 0);        // 14 floats (zeroed below)
    float*    W1f   = (float*)(ws + 64);       // 288*4 floats
    float*    b1f   = (float*)(ws + 4672);     // 288 floats
    _Float16* W2h   = (_Float16*)(ws + 5824);  // 288*288 halves (16B aligned)

    hipMemsetAsync(stats, 0, 64, stream);
    stats_kernel<<<512, 256, 0, stream>>>((const float4*)xyz, stats, N);
    prep_kernel<<<1, COUT, 0, stream>>>(W1, b1, gamma, beta, W2, stats, N, W1f, b1f, W2h);
    int blocks = (N + ROWS - 1) / ROWS;
    fused_kernel<<<blocks, COUT, 0, stream>>>(xyz, W1f, b1f, W2h, b2, (float*)d_out, N);
}
